// DSTAM_34935263985699
// MI455X (gfx1250) — matrix operand order repaired and checked by reading, not compiled, not run on hardware
//
#include <hip/hip_runtime.h>
#include <math.h>

// ---------------- constants for this problem instance ----------------
#define NN 2
#define TT 12
#define HH 96
#define WW 96
#define CC 64
#define H4 24
#define W4 24
#define LL 576          // H4*W4
#define FL 1024         // feat_len = c*16
#define GG 4
#define CGRP 256        // CG
#define ICN 192         // fusion conv input channels
#define KTOT 1728       // 192 ic * 9 taps (fusion GEMM K)
#define KSTEPS (KTOT / 32)
#define PIX 9216        // 96*96

typedef __attribute__((ext_vector_type(16))) _Float16 v16h;
typedef __attribute__((ext_vector_type(8)))  float    v8f;

__device__ __forceinline__ float wave_sum(float v) {
#pragma unroll
  for (int m = 16; m >= 1; m >>= 1) v += __shfl_xor(v, m, 32);
  return v;
}

// bilinear taps on a 24x24 image, reference semantics (validity on float
// coords BEFORE clipping, weight zeroed when invalid)
__device__ __forceinline__ void bilin_taps(float gx2, float gy2, int* ofs, float* wt) {
  float x = (gx2 + 1.f) * 0.5f * (float)(W4 - 1);
  float y = (gy2 + 1.f) * 0.5f * (float)(H4 - 1);
  float x0 = floorf(x), y0 = floorf(y);
  float wx1 = x - x0, wy1 = y - y0;
  float xs[2] = {x0, x0 + 1.f}, ys[2] = {y0, y0 + 1.f};
  float wxv[2] = {1.f - wx1, wx1}, wyv[2] = {1.f - wy1, wy1};
#pragma unroll
  for (int ty = 0; ty < 2; ++ty)
#pragma unroll
    for (int tx = 0; tx < 2; ++tx) {
      float xf = xs[tx], yf = ys[ty];
      bool valid = (xf >= 0.f) && (xf < (float)W4) && (yf >= 0.f) && (yf < (float)H4);
      int xc = min(max((int)xf, 0), W4 - 1);
      int yc = min(max((int)yf, 0), H4 - 1);
      ofs[ty * 2 + tx] = yc * W4 + xc;
      wt[ty * 2 + tx] = valid ? (wyv[ty] * wxv[tx]) : 0.f;
    }
}

// ---------------- K0: cf = l2norm(unfold(curr_feat)) : [n][l][1024] --------
__global__ void k_build_cf(const float* __restrict__ curr, float* __restrict__ cf) {
  int wid = (blockIdx.x * blockDim.x + threadIdx.x) >> 5;
  int lane = threadIdx.x & 31;
  if (wid >= NN * LL) return;
  int nidx = wid / LL, l = wid % LL;
  int y4 = l / W4, x4 = l % W4;
  float vals[32];
  float ss = 0.f;
#pragma unroll
  for (int i = 0; i < 32; ++i) {
    int cl = i * 32 + lane;                 // cl = c_idx*16 + sy*4 + sx
    int c_idx = cl >> 4, sy = (cl >> 2) & 3, sx = cl & 3;
    float v = curr[((size_t)(nidx * CC + c_idx) * HH + (y4 * 4 + sy)) * WW + (x4 * 4 + sx)];
    vals[i] = v;
    ss += v * v;
  }
  ss = wave_sum(ss);
  float inv = 1.f / fmaxf(sqrtf(ss), 1e-12f);
  size_t base = ((size_t)nidx * LL + l) * FL;
#pragma unroll
  for (int i = 0; i < 32; ++i) cf[base + i * 32 + lane] = vals[i] * inv;
}

// ------- K1: temporal_k = l2norm(nearest-sample(index_feat)) + nloc --------
// tk layout: [b=n*t][cl][l] ; nloc packed xc | yc<<8 | valid<<16 per [b][l]
__global__ void k_build_tk(const float* __restrict__ idxfeat, const float* __restrict__ locf,
                           float* __restrict__ tk, int* __restrict__ nloc) {
  int wid = (blockIdx.x * blockDim.x + threadIdx.x) >> 5;
  int lane = threadIdx.x & 31;
  if (wid >= NN * TT * LL) return;
  int b = wid / LL, l = wid % LL;
  int nidx = b / TT, tt = b % TT;
  int y4 = l / W4, x4 = l % W4;
  // sample coord == raw location value (normalize/denormalize cancels)
  float xf = locf[((size_t)(nidx * 2 * TT + tt * 2 + 0) * H4 + y4) * W4 + x4];
  float yf = locf[((size_t)(nidx * 2 * TT + tt * 2 + 1) * H4 + y4) * W4 + x4];
  int xi = (int)rintf(xf), yi = (int)rintf(yf);   // round-half-even like jnp.round
  bool valid = (xi >= 0 && xi < W4 && yi >= 0 && yi < H4);
  int xc = min(max(xi, 0), W4 - 1), yc = min(max(yi, 0), H4 - 1);
  if (lane == 0) nloc[(size_t)b * LL + l] = xc | (yc << 8) | (valid ? (1 << 16) : 0);
  const float* src = idxfeat + (size_t)b * FL * LL;
  int loc = yc * W4 + xc;
  float vals[32];
  float ss = 0.f;
#pragma unroll
  for (int i = 0; i < 32; ++i) {
    int cl = i * 32 + lane;
    float v = valid ? src[(size_t)cl * LL + loc] : 0.f;
    vals[i] = v;
    ss += v * v;
  }
  ss = wave_sum(ss);
  float inv = 1.f / fmaxf(sqrtf(ss), 1e-12f);
#pragma unroll
  for (int i = 0; i < 32; ++i)
    tk[((size_t)b * FL + i * 32 + lane) * LL + l] = vals[i] * inv;
}

// --- K2: grouped 5x5 conv + LN + GELU + 1x1 conv -> tanh offsets -> grid2 --
// one wave per (b96, pixel); lane owns 8 of the 256 channels
__global__ void k_offset(const float* __restrict__ cf, const float* __restrict__ tk,
                         const float* __restrict__ dww, const float* __restrict__ dwb,
                         const float* __restrict__ lnw, const float* __restrict__ lnb,
                         const float* __restrict__ pww, float* __restrict__ grid2) {
  int wid = (blockIdx.x * blockDim.x + threadIdx.x) >> 5;
  int lane = threadIdx.x & 31;
  if (wid >= NN * TT * GG * LL) return;
  int b = wid / LL, pix = wid % LL;
  int y = pix / W4, x = pix % W4;
  int m = b & 7;                       // q side: jnp.tile -> index b % (n*G)
  int qn = m >> 2, qg = m & 3;
  int kn = b / (TT * GG), ktt = (b / GG) % TT, kg = b & 3;
  const float* qbase = cf + (size_t)qn * LL * FL;                  // [pix][1024]
  const float* kbase = tk + (size_t)(kn * TT + ktt) * FL * LL;     // [cl][pix]
  float acc[8];
#pragma unroll
  for (int i = 0; i < 8; ++i) {
    int k = lane + 32 * i;             // output channel
    float s = dwb[k];
#pragma unroll
    for (int j = 0; j < 2; ++j) {
      int ic = 2 * k + j;              // group k uses input channels 2k,2k+1
      bool isq = ic < CGRP;
      int cg = isq ? ic : (ic - CGRP);
      for (int dy = 0; dy < 5; ++dy)
        for (int dx = 0; dx < 5; ++dx) {
          int yy = y + dy - 2, xx = x + dx - 2;
          if (yy < 0 || yy >= H4 || xx < 0 || xx >= W4) continue;
          float v = isq ? qbase[(size_t)(yy * W4 + xx) * FL + qg * CGRP + cg]
                        : kbase[(size_t)(kg * CGRP + cg) * LL + yy * W4 + xx];
          s += dww[((k * 2 + j) * 5 + dy) * 5 + dx] * v;
        }
    }
    acc[i] = s;
  }
  float sum = 0.f, sq = 0.f;
#pragma unroll
  for (int i = 0; i < 8; ++i) { sum += acc[i]; sq += acc[i] * acc[i]; }
  sum = wave_sum(sum);
  sq = wave_sum(sq);
  float mu = sum * (1.f / 256.f);
  float var = sq * (1.f / 256.f) - mu * mu;
  float rstd = rsqrtf(var + 1e-5f);
  float o0 = 0.f, o1 = 0.f;
#pragma unroll
  for (int i = 0; i < 8; ++i) {
    int k = lane + 32 * i;
    float tv = (acc[i] - mu) * rstd * lnw[k] + lnb[k];
    float g = 0.5f * tv * (1.f + erff(tv * 0.70710678118654752f));  // exact GELU
    o0 += pww[k] * g;
    o1 += pww[CGRP + k] * g;
  }
  o0 = wave_sum(o0);
  o1 = wave_sum(o1);
  if (lane == 0) {
    float off0 = tanhf(o0) * (2.f / (float)H4);  // channel 0 -> y component
    float off1 = tanhf(o1) * (2.f / (float)W4);  // channel 1 -> x component
    float ry = ((y + 0.5f) / (float)H4) * 2.f - 1.f;
    float rx = ((x + 0.5f) / (float)W4) * 2.f - 1.f;
    grid2[((size_t)b * LL + pix) * 2 + 0] = off1 + rx;   // x coord (pos reversed)
    grid2[((size_t)b * LL + pix) * 2 + 1] = off0 + ry;   // y coord
  }
}

// ---- K3: matrix_index = <bilerp(tk,grid2), cf> over 1024; max/argmax over t
__global__ void k_corr(const float* __restrict__ tk, const float* __restrict__ cf,
                       const float* __restrict__ grid2, float* __restrict__ csoft,
                       int* __restrict__ cidx) {
  int wid = (blockIdx.x * blockDim.x + threadIdx.x) >> 5;
  int lane = threadIdx.x & 31;
  if (wid >= NN * LL) return;
  int nidx = wid / LL, l = wid % LL;
  const float* cfl = cf + ((size_t)nidx * LL + l) * FL;
  float best = -3.4e38f;
  int bestt = 0;
  for (int tt = 0; tt < TT; ++tt) {
    float part = 0.f;
    const float* tkb = tk + (size_t)(nidx * TT + tt) * FL * LL;
#pragma unroll
    for (int g = 0; g < GG; ++g) {
      int b96 = nidx * (TT * GG) + tt * GG + g;
      float gx2 = grid2[((size_t)b96 * LL + l) * 2 + 0];
      float gy2 = grid2[((size_t)b96 * LL + l) * 2 + 1];
      int ofs[4];
      float wt[4];
      bilin_taps(gx2, gy2, ofs, wt);
#pragma unroll
      for (int i8 = 0; i8 < 8; ++i8) {
        int cl = g * CGRP + i8 * 32 + lane;
        const float* tp = tkb + (size_t)cl * LL;
        float v = wt[0] * tp[ofs[0]] + wt[1] * tp[ofs[1]] +
                  wt[2] * tp[ofs[2]] + wt[3] * tp[ofs[3]];
        part += v * cfl[cl];
      }
    }
    part = wave_sum(part);
    if (part > best) { best = part; bestt = tt; }   // first max kept (strict >)
  }
  if (lane == 0) {
    csoft[(size_t)nidx * LL + l] = best;
    cidx[(size_t)nidx * LL + l] = bestt;
  }
}

// ---- K4: sample_gather for s1/s2/s3 (nearest composed with bilinear) + fold
// writes folded features as f16 [n][192][96][96]
__global__ void k_gather(const float* __restrict__ s1, const float* __restrict__ s2,
                         const float* __restrict__ s3, const float* __restrict__ grid2,
                         const int* __restrict__ nloc, const int* __restrict__ cidx,
                         _Float16* __restrict__ fin) {
  size_t tid = (size_t)blockIdx.x * blockDim.x + threadIdx.x;
  const size_t total = (size_t)NN * 3 * FL * LL;
  if (tid >= total) return;
  int l = (int)(tid % LL);
  int cl = (int)((tid / LL) % FL);
  int s = (int)((tid / ((size_t)LL * FL)) % 3);
  int nidx = (int)(tid / ((size_t)LL * FL * 3));
  int tt = cidx[(size_t)nidx * LL + l];
  int g = cl >> 8;
  int b96 = nidx * (TT * GG) + tt * GG + g;
  float gx2 = grid2[((size_t)b96 * LL + l) * 2 + 0];
  float gy2 = grid2[((size_t)b96 * LL + l) * 2 + 1];
  int ofs[4];
  float wt[4];
  bilin_taps(gx2, gy2, ofs, wt);
  const float* sp = (s == 0) ? s1 : (s == 1) ? s2 : s3;
  const float* src = sp + ((size_t)(nidx * TT + tt) * FL + cl) * LL;
  const int* nl = nloc + (size_t)(nidx * TT + tt) * LL;
  float v = 0.f;
#pragma unroll
  for (int tpi = 0; tpi < 4; ++tpi) {
    if (wt[tpi] != 0.f) {
      int packed = nl[ofs[tpi]];
      if (packed >> 16) v += wt[tpi] * src[((packed >> 8) & 255) * W4 + (packed & 255)];
    }
  }
  int c_idx = cl >> 4, sy = (cl >> 2) & 3, sx = cl & 3;
  int y4 = l / W4, x4 = l % W4;
  fin[((size_t)(nidx * ICN + s * 64 + c_idx)) * PIX + (y4 * 4 + sy) * WW + (x4 * 4 + sx)] =
      (_Float16)v;
}

// ---- K5a: fusion weights (64,192,3,3) -> f16 flat [oc][K=1728] ------------
__global__ void k_wcvt(const float* __restrict__ w, _Float16* __restrict__ wh) {
  int i = blockIdx.x * blockDim.x + threadIdx.x;
  if (i < 64 * KTOT) wh[i] = (_Float16)w[i];
}

// ---- K5: fusion 3x3 conv 192->64 as implicit GEMM via WMMA f32<=f16 -------
// block = 4 waves, one 16-pixel tile (tiles never cross rows: 96 = 6*16).
// B operands are PRE-SWIZZLED into WMMA fragment layout in LDS by a one-time
// cooperative fill (all im2col index math + border checks paid once). The
// steady-state K-loop per step is branch-free and pack-free:
//   2x global_load_b128 (A weights) + 2x ds_load_b128 (B frag) + 1x v_wmma.
__global__ void __launch_bounds__(128) k_fusion(
    const _Float16* __restrict__ fin, const _Float16* __restrict__ wh,
    const float* __restrict__ fb, const float* __restrict__ csoft,
    const float* __restrict__ anchor, float* __restrict__ out) {
  __shared__ _Float16 bfrag[KSTEPS * 512];      // 54*512 f16 = 55296 B
  int bid = blockIdx.x;                         // n(2) x pixel-tile(576)
  int nidx = bid / (PIX / 16);
  int pt = bid % (PIX / 16);
  int py0 = pt / 6;                             // tile row
  int px0 = (pt % 6) * 16;                      // tile first column
  const _Float16* finb = fin + (size_t)nidx * ICN * PIX;

  // one-time fill: bfrag[step][lane][slot] = exact v16h contents per lane.
  // slot s of lane ln corresponds to K = step*32 + (s<8?0:16) + (ln<16?0:8) + (s&7), which
  // decodes to input channel ic = K/9 and 3x3 tap (ky,kx) = (K%9)/3, (K%9)%3.
  for (int idx = threadIdx.x; idx < KSTEPS * 512; idx += 128) {
    int step = idx >> 9;
    int r = idx & 511;
    int ln = r >> 4;                            // destination lane 0..31
    int s = r & 15;                             // slot 0..15
    int kf = step * 32 + ((s < 8) ? 0 : 16) + ((ln < 16) ? 0 : 8) + (s & 7);
    int ic = kf / 9;
    int tap = kf - ic * 9;
    int ky = tap / 3, kx = tap - ky * 3;
    int jn = ln & 15;                           // pixel within tile
    int yy = py0 + ky - 1, xx = px0 + jn + kx - 1;
    _Float16 v = (_Float16)0.f;
    if (yy >= 0 && yy < HH && xx >= 0 && xx < WW)
      v = finb[(size_t)ic * PIX + yy * WW + xx];
    bfrag[idx] = v;
  }
  __syncthreads();

  int wave = threadIdx.x >> 5;
  int lane = threadIdx.x & 31;
  int ocbase = wave * 16;
  int jn = lane & 15;                           // N (pixel within tile)
  int oc_a = ocbase + jn;                       // A: M = lane&15
  int kaHi = (lane < 16) ? 0 : 8;               // A K-half select (ISA layout)
  const _Float16* wrow = wh + (size_t)oc_a * KTOT;
  v8f acc = {};
  for (int step = 0; step < KSTEPS; ++step) {
    int ks = step * 32;
    v16h a;
    // A (16x32 f16): VGPR v half e -> K = ks + (v<4?0:16) + kaHi + (v&3)*2 + e
    // -> two contiguous 16B chunks per lane => 2x global_load_b128
#pragma unroll
    for (int v = 0; v < 8; ++v) {
      int kb = (v < 4 ? 0 : 16) + kaHi + (v & 3) * 2;
      a[2 * v + 0] = wrow[ks + kb + 0];
      a[2 * v + 1] = wrow[ks + kb + 1];
    }
    // B: 32 contiguous, 32B-aligned bytes per lane => 2x ds_load_b128
    v16h bm = *(const v16h*)&bfrag[step * 512 + lane * 16];
    acc = __builtin_amdgcn_wmma_f32_16x16x32_f16(false, a, false, bm, (short)0, acc,
                                                 false, false);
  }
  // D layout: VGPR r, lane -> M = r + (lane<16?0:8), N = lane&15
  int p = pt * 16 + jn;
  int py = p / WW, px = p % WW;
  int l = (py >> 2) * W4 + (px >> 2);
  float cs = csoft[(size_t)nidx * LL + l];
#pragma unroll
  for (int r = 0; r < 8; ++r) {
    int oc = ocbase + r + ((lane < 16) ? 0 : 8);
    size_t oidx = ((size_t)(nidx * CC + oc)) * PIX + py * WW + px;
    out[oidx] = (acc[r] + fb[oc]) * cs + anchor[oidx];
  }
}

// --------------------------------------------------------------------------
extern "C" void kernel_launch(void* const* d_in, const int* in_sizes, int n_in,
                              void* d_out, int out_size, void* d_ws, size_t ws_size,
                              hipStream_t stream) {
  const float* curr   = (const float*)d_in[0];
  const float* idxf   = (const float*)d_in[1];
  const float* anchor = (const float*)d_in[2];
  const float* s1     = (const float*)d_in[3];
  const float* s2     = (const float*)d_in[4];
  const float* s3     = (const float*)d_in[5];
  const float* locf   = (const float*)d_in[6];
  const float* fw     = (const float*)d_in[7];
  const float* fb     = (const float*)d_in[8];
  const float* dww    = (const float*)d_in[9];
  const float* dwb    = (const float*)d_in[10];
  const float* lnw    = (const float*)d_in[11];
  const float* lnb    = (const float*)d_in[12];
  const float* pww    = (const float*)d_in[13];
  float* out = (float*)d_out;

  char* base = (char*)d_ws;
  size_t off = 0;
  auto carve = [&](size_t bytes) {
    char* p = base + off;
    off = (off + bytes + 255) & ~(size_t)255;
    return p;
  };
  float*    cf    = (float*)carve((size_t)NN * LL * FL * 4);           // 4.7 MB
  float*    tk    = (float*)carve((size_t)NN * TT * FL * LL * 4);      // 56.6 MB
  int*      nloc  = (int*)carve((size_t)NN * TT * LL * 4);
  float*    grid2 = (float*)carve((size_t)NN * TT * GG * LL * 2 * 4);
  float*    csoft = (float*)carve((size_t)NN * LL * 4);
  int*      cidx  = (int*)carve((size_t)NN * LL * 4);
  _Float16* fin   = (_Float16*)carve((size_t)NN * ICN * PIX * 2);      // 7.1 MB
  _Float16* wh    = (_Float16*)carve((size_t)64 * KTOT * 2);           // 216 KB
  (void)in_sizes; (void)n_in; (void)out_size; (void)ws_size;

  k_build_cf<<<(NN * LL) / 8, 256, 0, stream>>>(curr, cf);
  k_build_tk<<<(NN * TT * LL) / 8, 256, 0, stream>>>(idxf, locf, tk, nloc);
  k_wcvt<<<(64 * KTOT + 255) / 256, 256, 0, stream>>>(fw, wh);
  k_offset<<<(NN * TT * GG * LL) / 8, 256, 0, stream>>>(cf, tk, dww, dwb, lnw, lnb,
                                                        pww, grid2);
  k_corr<<<(NN * LL) / 8, 256, 0, stream>>>(tk, cf, grid2, csoft, cidx);
  k_gather<<<(int)(((size_t)NN * 3 * FL * LL) / 256), 256, 0, stream>>>(
      s1, s2, s3, grid2, nloc, cidx, fin);
  k_fusion<<<NN * (PIX / 16), 128, 0, stream>>>(fin, wh, fb, csoft, anchor, out);
}
